// WindowAttention_71614284693731
// MI455X (gfx1250) — matrix operand order repaired and checked by reading, not compiled, not run on hardware
//
#include <hip/hip_runtime.h>
#include <hip/hip_bf16.h>

// ---------------------------------------------------------------------------
// Swin-style shifted window attention on MI455X (gfx1250), bf16 WMMA pipeline.
//   b=8, L=4096, dim=512, H=8, HD=64, WS=64, n=64 windows, DISP=32
// Memory-bound (~73 GFLOP vs ~400MB @ 23.3TB/s). All matmuls lower to
// v_wmma_f32_16x16x32_bf16. Tile staging uses the CDNA5 async-to-LDS path
// (global_load_async_to_lds_b128 + s_wait_asynccnt) with double buffering so
// tile i+1's DMA overlaps tile i's WMMAs and data never touches VGPRs.
// ---------------------------------------------------------------------------

typedef __attribute__((ext_vector_type(16))) __bf16 v16bf;
typedef __attribute__((ext_vector_type(8)))  float  v8f;

union FragU { unsigned short u[16]; v16bf v; };

__device__ __forceinline__ unsigned short f2b(float f) {
  unsigned int u = __float_as_uint(f);
  u += 0x7FFFu + ((u >> 16) & 1u);      // round-to-nearest-even
  return (unsigned short)(u >> 16);
}
__device__ __forceinline__ float b2f(unsigned short h) {
  return __uint_as_float(((unsigned int)h) << 16);
}

// --- CDNA5 async global->LDS copy (ASYNCcnt-tracked, bypasses VGPRs) -------
// LDS aperture: generic pointer's low 32 bits are the wave-relative LDS
// address (ISA 10.2), which is what VDST of the async op consumes.
__device__ __forceinline__ void async_ld_b128(void* lds, const void* g) {
  const unsigned int laddr = (unsigned int)(size_t)lds;
  asm volatile("global_load_async_to_lds_b128 %0, %1, off"
               :: "v"(laddr), "v"(g) : "memory");
}
__device__ __forceinline__ void wait_async0() {   // all async copies landed
  asm volatile("s_wait_asynccnt 0x0" ::: "memory");
}
__device__ __forceinline__ void wait_async4() {   // oldest tile (4/thr) landed
  asm volatile("s_wait_asynccnt 0x4" ::: "memory");
}

// 16-bit A-matrix 16x32 lane mapping used here (matches loadBrow):
//  VGPR p : K = g*16 + 2p + e   (g = lane/16)
// Per lane: one contiguous 16-element run -> 2x ds_load_b128.
__device__ __forceinline__ int fragAK(int p, int e, int g) {
  return ((p < 4) ? (g * 8 + 2 * p) : (16 + g * 8 + 2 * (p - 4))) + e;
}

// A fragment (16x32 bf16): rows m0..m0+15, K = k0..k0+31 from LDS [m][k], ld=lda
__device__ __forceinline__ v16bf loadA(const unsigned short* s, int lda, int m0,
                                       int k0, int lane) {
  FragU t; const int m = m0 + (lane & 15); const int g = lane >> 4;
#pragma unroll
  for (int p = 0; p < 8; ++p)
#pragma unroll
    for (int e = 0; e < 2; ++e)
      t.u[2 * p + e] = s[m * lda + k0 + g * 16 + 2 * p + e];
  return t.v;
}

// B fragment (32x16 bf16, B[k][n]): lane holds column n = n0 + lane&15;
// lanes 0-15 pack K = k0..k0+15, lanes 16-31 pack K = k0+16..k0+31.
// Source LDS stored K-contiguous per column: S[n][k], ld = lds
// -> each lane reads 16 contiguous shorts (32B) = 2x ds_load_b128.
__device__ __forceinline__ v16bf loadBrow(const unsigned short* s, int lds,
                                          int k0, int n0, int lane) {
  FragU t; const int n = n0 + (lane & 15); const int g = lane >> 4;
#pragma unroll
  for (int e = 0; e < 16; ++e)
    t.u[e] = s[n * lds + k0 + g * 16 + e];
  return t.v;
}

__device__ __forceinline__ v8f wmma_bf16(v16bf a, v16bf b, v8f c) {
  return __builtin_amdgcn_wmma_f32_16x16x32_bf16(false, a, false, b, (short)0, c,
                                                 false, false);
}

// ---------------------------------------------------------------------------
// f32 -> bf16 conversions (vectorized 4/thread), x with -DISP roll fused.
// ---------------------------------------------------------------------------
__global__ __launch_bounds__(256) void conv_shift_x(const float* __restrict__ x,
                                                    unsigned short* __restrict__ o) {
  const int q = blockIdx.x * 256 + threadIdx.x;             // 4,194,304 quads
  const int idx = q * 4;                                    // 16,777,216 elems
  const int b   = idx >> 21;
  const int rem = idx & ((1 << 21) - 1);
  const int r = rem >> 9, c = rem & 511;                    // c multiple of 4
  const float4 v = *(const float4*)(x + ((size_t)b << 21) +
                                    ((size_t)((r + 32) & 4095) << 9) + c);
  ushort4 p; p.x = f2b(v.x); p.y = f2b(v.y); p.z = f2b(v.z); p.w = f2b(v.w);
  *(ushort4*)(o + idx) = p;
}

__global__ __launch_bounds__(256) void conv_f2b(const float* __restrict__ in,
                                                unsigned short* __restrict__ out,
                                                int nQuad) {
  const int q = blockIdx.x * 256 + threadIdx.x;
  if (q < nQuad) {
    const float4 v = *(const float4*)(in + q * 4);
    ushort4 p; p.x = f2b(v.x); p.y = f2b(v.y); p.z = f2b(v.z); p.w = f2b(v.w);
    *(ushort4*)(out + q * 4) = p;
  }
}

// ---------------------------------------------------------------------------
// Tiled bf16 WMMA GEMM:  C[M,N] = A[M,K] * W[N,K]^T   (B[k][n] = W[n][k])
// 64x64 tile per 128-thread workgroup (4 waves); each wave owns a 16-row strip.
// Double-buffered K loop; tiles DMA'd into LDS with global_load_async_to_lds.
// Both LDS tiles stored K-contiguous ([m][k], [n][k]) so B staging is a pure
// contiguous copy from W and all fragment loads vectorize to ds_load_b128.
// mode 0: bf16 store to Cb. mode 1: f32 store to Cf + bias + fused +DISP roll.
// ---------------------------------------------------------------------------
__global__ __launch_bounds__(128)
void gemm_bf16_wmma(const unsigned short* __restrict__ A,
                    const unsigned short* __restrict__ W,
                    unsigned short* __restrict__ Cb,
                    float* __restrict__ Cf,
                    const float* __restrict__ bias,
                    int M, int N, int K, int mode) {
  constexpr int LDT = 40;                                   // 32 + pad (shorts)
  constexpr int TS  = 64 * LDT;                             // one tile (shorts)
  __shared__ __align__(16) unsigned short As[2 * TS];       // [buf][m][k]
  __shared__ __align__(16) unsigned short Bs[2 * TS];       // [buf][n][k]
  const int t = threadIdx.x, lane = t & 31, wid = t >> 5;
  const int n0 = blockIdx.x * 64, mB = blockIdx.y * 64;
  const int m0 = wid * 16;
  const int r = t >> 1, cb = (t & 1) * 16;                  // staging coords
  v8f acc[4] = {};

  const unsigned short* gA = A + (size_t)(mB + r) * K + cb; // thread's A row
  const unsigned short* gB = W + (size_t)(n0 + r) * K + cb; // thread's W row

  // issue one 64x32 A tile + B tile into buffer `buf` (4 async b128 / thread)
  auto issue_tile = [&](int k0, int buf) {
    unsigned short* la = As + buf * TS + r * LDT + cb;
    unsigned short* lb = Bs + buf * TS + r * LDT + cb;
    async_ld_b128(la,     gA + k0);
    async_ld_b128(la + 8, gA + k0 + 8);
    async_ld_b128(lb,     gB + k0);
    async_ld_b128(lb + 8, gB + k0 + 8);
  };

  const int nk = K >> 5;                                    // K / 32 steps
  issue_tile(0, 0);
  for (int i = 0; i < nk; ++i) {
    const int cur = i & 1;
    if (i + 1 < nk) {
      issue_tile((i + 1) << 5, cur ^ 1);                    // prefetch next
      wait_async4();                                        // oldest 4 landed
    } else {
      wait_async0();
    }
    __syncthreads();                                        // tile visible WG-wide

    const unsigned short* as = As + cur * TS;
    const unsigned short* bs = Bs + cur * TS;
    const v16bf a = loadA(as, LDT, m0, 0, lane);            // reused 4x
#pragma unroll
    for (int j = 0; j < 4; ++j) {
      const v16bf b = loadBrow(bs, LDT, 0, j * 16, lane);
      acc[j] = wmma_bf16(a, b, acc[j]);
    }
    __syncthreads();                                        // done before overwrite
  }

  // C/D layout: VGPR rr -> row m0+rr+8*(lane/16), col = n-tile + lane%16
  const int col = lane & 15, g = lane >> 4;
#pragma unroll
  for (int j = 0; j < 4; ++j)
#pragma unroll
    for (int rr2 = 0; rr2 < 8; ++rr2) {
      const int row = mB + m0 + rr2 + 8 * g;
      const int c = n0 + j * 16 + col;
      float v = acc[j][rr2];
      if (mode == 0) {
        Cb[(size_t)row * N + c] = f2b(v);
      } else {
        v += bias[c];
        const int bi = row >> 12, rm = row & 4095;          // fused +DISP roll
        const int orow = (bi << 12) | ((rm + 32) & 4095);
        Cf[(size_t)orow * N + c] = v;
      }
    }
}

// ---------------------------------------------------------------------------
// k_sum[b,h,j,d] = sum_n k[b,h,n,j,d]   (k = qkv cols 512..1023), bf16 out.
// ---------------------------------------------------------------------------
__global__ __launch_bounds__(256)
void ksum_kernel(const unsigned short* __restrict__ qkv,
                 unsigned short* __restrict__ ks) {
  const int bh = blockIdx.x;                                // 0..63
  const int b = bh >> 3, h = bh & 7;
  for (int idx = threadIdx.x; idx < 64 * 64; idx += 256) {
    const int j = idx >> 6, d = idx & 63;
    const unsigned short* p =
        qkv + ((size_t)b * 4096 + j) * 1536 + 512 + h * 64 + d;
    float s = 0.f;
#pragma unroll 4
    for (int n = 0; n < 64; ++n) {
      if ((n & 7) == 0 && n + 8 < 64)                       // global_prefetch_b8
        __builtin_prefetch(p + (size_t)(n + 8) * 64 * 1536, 0, 0);
      s += b2f(p[(size_t)n * 64 * 1536]);
    }
    ks[(size_t)bh * 4096 + idx] = f2b(s);
  }
}

// ---------------------------------------------------------------------------
// Per-(b,h,n) window attention: dots = q*k_sum^T*scale + pos[63-j] + mask,
// masked softmax over the valid 32-column half, out = attn*v. All WMMA.
// grid = (n=64, h=8, b=8), 128 threads (4 waves).
// LDS layouts chosen K-contiguous for every fragment:
//   Qs[i][d] (A of pass1; attn[i][j] A of pass2 after softmax overwrite)
//   Ks[j][d] (B of pass1: B[k=d][n=j] -> lane reads row j contiguously)
//   Vt[d][j] (B of pass2: B[k=j][n=d] -> lane reads row d contiguously)
// q/k_sum staged via async DMA, overlapping the synchronous V transpose.
// ---------------------------------------------------------------------------
__global__ __launch_bounds__(128)
void win_attn_kernel(const unsigned short* __restrict__ qkv,
                     const unsigned short* __restrict__ ksum,
                     const float* __restrict__ pos,
                     unsigned short* __restrict__ ao) {
  constexpr int LQ = 72, LK = 72, LV = 72, LD = 66;
  __shared__ __align__(16) unsigned short Qs[64 * LQ];      // q, then attn(bf16)
  __shared__ __align__(16) unsigned short Ks[64 * LK];      // k_sum [j][d]
  __shared__ __align__(16) unsigned short Vt[64 * LV];      // v^T   [d][j]
  __shared__ float Ds[64 * LD];                             // dots f32

  const int n = blockIdx.x, h = blockIdx.y, b = blockIdx.z;
  const int t = threadIdx.x, lane = t & 31, wid = t >> 5;
  const size_t rowBase = ((size_t)b * 4096 + (size_t)n * 64) * 1536 + h * 64;

  { // stage: q + k_sum via async DMA, v transposed synchronously (overlap)
    const int r = t >> 1, db = (t & 1) * 32;
    const unsigned short* gq = qkv + rowBase + (size_t)r * 1536 + db;
    const unsigned short* gk = ksum + ((size_t)(b * 8 + h) * 64 + r) * 64 + db;
    unsigned short* lq = Qs + r * LQ + db;
    unsigned short* lk = Ks + r * LK + db;
#pragma unroll
    for (int i = 0; i < 4; ++i) {                           // 4x 16B each
      async_ld_b128(lq + i * 8, gq + i * 8);
      async_ld_b128(lk + i * 8, gk + i * 8);
    }
    // v[j][d] -> Vt[d][j]; thread owns row j=r, 32 d values starting at db
    unsigned short vbuf[32];
    const uint4* v = (const uint4*)(qkv + rowBase + (size_t)r * 1536 + 1024 + db);
    *(uint4*)&vbuf[0] = v[0]; *(uint4*)&vbuf[8]  = v[1];
    *(uint4*)&vbuf[16] = v[2]; *(uint4*)&vbuf[24] = v[3];
#pragma unroll
    for (int i = 0; i < 32; ++i) Vt[(db + i) * LV + r] = vbuf[i];
    wait_async0();
  }
  __syncthreads();

  const int m0 = wid * 16;
  // ---- pass 1: dots = q @ k_sum^T -----------------------------------------
  v8f acc[4] = {};
#pragma unroll
  for (int k0 = 0; k0 < 64; k0 += 32) {
    const v16bf a = loadA(Qs, LQ, m0, k0, lane);
#pragma unroll
    for (int j = 0; j < 4; ++j) {
      const v16bf bm = loadBrow(Ks, LK, k0, j * 16, lane);  // B[d][j] = Ks[j][d]
      acc[j] = wmma_bf16(a, bm, acc[j]);
    }
  }
  {
    const float scale = 0.125f;                             // 64^-0.5
    const int col = lane & 15, g = lane >> 4;
#pragma unroll
    for (int j = 0; j < 4; ++j)
#pragma unroll
      for (int rr = 0; rr < 8; ++rr)
        Ds[(m0 + rr + 8 * g) * LD + j * 16 + col] = acc[j][rr] * scale;
  }
  __syncthreads();

  // ---- masked softmax: -inf outside the row's 32-col half -----------------
  if (t < 64) {
    const int row = t;
    const int jlo = (row < 32) ? 0 : 32;
    float xv[32];
    float m = -1e30f;
#pragma unroll
    for (int jj = 0; jj < 32; ++jj) {
      xv[jj] = Ds[row * LD + jlo + jj] + pos[63 - (jlo + jj)];
      m = fmaxf(m, xv[jj]);
    }
    float s = 0.f;
#pragma unroll
    for (int jj = 0; jj < 32; ++jj) { xv[jj] = __expf(xv[jj] - m); s += xv[jj]; }
    const float inv = 1.f / s;
#pragma unroll
    for (int c = 0; c < 64; ++c) {
      const float v = (c >= jlo && c < jlo + 32) ? xv[c - jlo] * inv : 0.f;
      Qs[row * LQ + c] = f2b(v);                            // attn (bf16) in Qs
    }
  }
  __syncthreads();

  // ---- pass 2: out = attn @ v ---------------------------------------------
  v8f oacc[4] = {};
#pragma unroll
  for (int k0 = 0; k0 < 64; k0 += 32) {
    const v16bf a = loadA(Qs, LQ, m0, k0, lane);
#pragma unroll
    for (int j = 0; j < 4; ++j) {
      const v16bf bm = loadBrow(Vt, LV, k0, j * 16, lane);  // B[j][d] = Vt[d][j]
      oacc[j] = wmma_bf16(a, bm, oacc[j]);
    }
  }
  {
    const int col = lane & 15, g = lane >> 4;
#pragma unroll
    for (int j = 0; j < 4; ++j)
#pragma unroll
      for (int rr = 0; rr < 8; ++rr) {
        const size_t row = (size_t)b * 4096 + (size_t)n * 64 + m0 + rr + 8 * g;
        ao[row * 512 + h * 64 + j * 16 + col] = f2b(oacc[j][rr]);
      }
  }
}

// ---------------------------------------------------------------------------
extern "C" void kernel_launch(void* const* d_in, const int* in_sizes, int n_in,
                              void* d_out, int out_size, void* d_ws, size_t ws_size,
                              hipStream_t stream) {
  const float* x    = (const float*)d_in[0];   // (8, 4096, 512)
  const float* wqkv = (const float*)d_in[1];   // (1536, 512)
  const float* wout = (const float*)d_in[2];   // (512, 512)
  const float* bout = (const float*)d_in[3];   // (512,)
  const float* pos  = (const float*)d_in[4];   // (127,)
  float* out = (float*)d_out;                  // (8, 4096, 512) f32

  // workspace layout (bf16 buffers, ~170 MB total)
  unsigned short* xb   = (unsigned short*)d_ws;               // 16,777,216
  unsigned short* wqb  = xb  + (size_t)16777216;              //    786,432
  unsigned short* wob  = wqb + (size_t)786432;                //    262,144
  unsigned short* ksb  = wob + (size_t)262144;                //    262,144
  unsigned short* aob  = ksb + (size_t)262144;                // 16,777,216
  unsigned short* qkvb = aob + (size_t)16777216;              // 50,331,648

  // 1) convert inputs to bf16 (x with -DISP roll fused)
  conv_shift_x<<<16777216 / 4 / 256, 256, 0, stream>>>(x, xb);
  conv_f2b<<<(786432 / 4) / 256, 256, 0, stream>>>(wqkv, wqb, 786432 / 4);
  conv_f2b<<<(262144 / 4) / 256, 256, 0, stream>>>(wout, wob, 262144 / 4);

  // 2) qkv = x_shift @ w_qkv^T   (32768 x 1536, K=512) -> bf16
  gemm_bf16_wmma<<<dim3(1536 / 64, 32768 / 64), 128, 0, stream>>>(
      xb, wqb, qkvb, nullptr, nullptr, 32768, 1536, 512, 0);

  // 3) k_sum over windows per (b,h)
  ksum_kernel<<<64, 256, 0, stream>>>(qkvb, ksb);

  // 4) window attention, one WG per (b,h,n)
  win_attn_kernel<<<dim3(64, 8, 8), 128, 0, stream>>>(qkvb, ksb, pos, aob);

  // 5) out = attn_out @ w_out^T + b_out, +DISP roll fused in the store
  gemm_bf16_wmma<<<dim3(512 / 64, 32768 / 64), 128, 0, stream>>>(
      aob, wob, nullptr, out, bout, 32768, 512, 512, 1);
}
